// SimpleRNN_29566554866196
// MI455X (gfx1250) — matrix operand order repaired and checked by reading, not compiled, not run on hardware
//
#include <hip/hip_runtime.h>
#include <hip/hip_bf16.h>
#include <math.h>

// ---------------------------------------------------------------------------
// SimpleRNN on MI455X (gfx1250, wave32, WMMA)
//   h = tanh([x_t | h] @ W_ih^T + b_ih), scanned 1024 steps in one persistent
//   kernel with a grid-wide barrier per step. Each WG stages its 16-column
//   W_ih slice (48 KB) into LDS once; per-step B fragments are ds_load_b128
//   (no L2 round-trip, no per-wave duplication). bf16 in / f32 accumulate via
//   v_wmma_f32_16x16x32_bf16.
// ---------------------------------------------------------------------------

typedef __bf16 bf16_t;
typedef __attribute__((ext_vector_type(16))) __bf16 v16bf;
typedef __attribute__((ext_vector_type(8)))  __bf16 v8bf;
typedef __attribute__((ext_vector_type(8)))  float  v8f;

#define BATCH   128
#define TSTEPS  1024
#define INSZ    512
#define HID     1024
#define OUTSZ   512
#define KTOT    (INSZ + HID)   // 1536
#define NWG     64
#define WAVES   4
#define KF_X    (INSZ / 32)    // 16 fragments over the x part
#define KF_H    (HID / 32)     // 32 fragments over the h part
#define KF_TOT  (KF_X + KF_H)  // 48

// ---- WMMA fragment loaders (contiguous k = 16h+s order in both A and B, wave32) ----------

// B fragment (32x16, bf16): lane L holds column n = n0+(L&15), 16 contiguous
// K values starting at k0 (+16 for upper lane half). Source is row-major
// (N rows x K cols), i.e. W[n*ldb + k] -- one 32-byte load per lane.
__device__ __forceinline__ v16bf load_b_frag(const bf16_t* B,
                                             int ldb, int n0, int k0, int lane) {
  int n = n0 + (lane & 15);
  int k = k0 + ((lane & 16) ? 16 : 0);
  return *(const v16bf*)(B + (size_t)n * ldb + k);
}

// A fragment (16x32, bf16): lane L holds row m = m0+(L&15); lower lane half
// holds K = k0+{0..15}, upper half K = k0+{16..31}.
__device__ __forceinline__ v16bf load_a_frag_bf16(const bf16_t* __restrict__ A,
                                                  int lda, int m0, int k0, int lane) {
  int m = m0 + (lane & 15);
  int k = k0 + ((lane & 16) ? 16 : 0);
  const bf16_t* p = A + (size_t)m * lda + k;
  v8bf lo = *(const v8bf*)(p);
  v8bf hi = *(const v8bf*)(p + 8);
  v16bf r;
#pragma unroll
  for (int i = 0; i < 8; ++i) { r[i] = lo[i]; r[i + 8] = hi[i]; }
  return r;
}

// A fragment sourced from fp32 x (convert to bf16 in-register).
__device__ __forceinline__ v16bf load_a_frag_f32(const float* __restrict__ A,
                                                 size_t lda, int m0, int k0, int lane) {
  int m = m0 + (lane & 15);
  int k = k0 + ((lane & 16) ? 16 : 0);
  const float* p = A + (size_t)m * lda + k;
  typedef __attribute__((ext_vector_type(8))) float f8;
  f8 lo = *(const f8*)(p);
  f8 hi = *(const f8*)(p + 8);
  v16bf r;
#pragma unroll
  for (int i = 0; i < 8; ++i) { r[i] = (__bf16)lo[i]; r[i + 8] = (__bf16)hi[i]; }
  return r;
}

__device__ __forceinline__ v8f wmma_bf16(v16bf a, v16bf b, v8f c) {
  return __builtin_amdgcn_wmma_f32_16x16x32_bf16(false, a, false, b,
                                                 (short)0, c, false, false);
}

// ---- prologue: fp32 -> bf16 weight copies + zero h state -------------------

__global__ void rnn_prologue(const float* __restrict__ Wih_f,
                             const float* __restrict__ Who_f,
                             bf16_t* __restrict__ Wih,
                             bf16_t* __restrict__ Who,
                             bf16_t* __restrict__ hbuf) {
  const int NW = HID * KTOT;          // 1,572,864
  const int NO = OUTSZ * HID;         //   524,288
  const int NH = 2 * BATCH * HID;     //   262,144
  const int total = NW + NO + NH;
  for (int i = blockIdx.x * blockDim.x + threadIdx.x; i < total;
       i += gridDim.x * blockDim.x) {
    if (i < NW)            Wih[i]          = (__bf16)Wih_f[i];
    else if (i < NW + NO)  Who[i - NW]     = (__bf16)Who_f[i - NW];
    else                   hbuf[i - NW - NO] = (__bf16)0.0f;
  }
}

// ---- persistent scan kernel ------------------------------------------------

__global__ void __launch_bounds__(WAVES * 32, 1)
rnn_persistent(const float* __restrict__ x,       // (B, T, INSZ) fp32
               const float* __restrict__ b_ih,    // (HID,)
               const float* __restrict__ b_ho,    // (OUTSZ,)
               const bf16_t* __restrict__ Wih,    // (HID, KTOT) bf16
               const bf16_t* __restrict__ Who,    // (OUTSZ, HID) bf16
               bf16_t* __restrict__ hbuf,         // 2 x (B, HID) bf16
               float* __restrict__ out,           // (B, OUTSZ) fp32
               unsigned* __restrict__ counter) {
  const int lane = threadIdx.x & 31;
  const int wave = threadIdx.x >> 5;
  const int wg   = blockIdx.x;

  const int n0  = wg * 16;                 // hidden columns owned by this WG
  const int m0a = (wave * 2 + 0) * 16;     // two batch-row tiles per wave
  const int m0b = (wave * 2 + 1) * 16;

  bf16_t* h0 = hbuf;
  bf16_t* h1 = hbuf + BATCH * HID;

  const float bias_h = b_ih[n0 + (lane & 15)];   // invariant across t
  const size_t x_lda = (size_t)TSTEPS * INSZ;    // row stride over batch dim

  // --- Stage this WG's 16-column W_ih slice into LDS (48 KB, one memcpy). ---
  // Rows n0..n0+15 of the row-major weight matrix are contiguous in memory.
  __shared__ bf16_t lds_w[16 * KTOT];
  {
    const bf16_t* src = Wih + (size_t)n0 * KTOT;
    for (int c = threadIdx.x; c < (16 * KTOT) / 8; c += WAVES * 32) {
      *(v8bf*)(lds_w + c * 8) = *(const v8bf*)(src + c * 8);
    }
  }
  __syncthreads();

  for (int t = 0; t < TSTEPS; ++t) {
    const bf16_t* hin  = (t & 1) ? h1 : h0;
    bf16_t*       hout = (t & 1) ? h0 : h1;
    const float*  xt   = x + (size_t)t * INSZ;

    v8f acc0 = {};
    v8f acc1 = {};

    // K = [0, 512): input projection (x is fp32, convert in-register)
#pragma unroll
    for (int kk = 0; kk < KF_X; ++kk) {
      v16bf b  = load_b_frag(lds_w, KTOT, 0, kk * 32, lane);
      v16bf a0 = load_a_frag_f32(xt, x_lda, m0a, kk * 32, lane);
      v16bf a1 = load_a_frag_f32(xt, x_lda, m0b, kk * 32, lane);
      acc0 = wmma_bf16(a0, b, acc0);
      acc1 = wmma_bf16(a1, b, acc1);
    }
    // K = [512, 1536): recurrent part (h is bf16, L2 resident)
#pragma unroll
    for (int kk = 0; kk < KF_H; ++kk) {
      v16bf b  = load_b_frag(lds_w, KTOT, 0, INSZ + kk * 32, lane);
      v16bf a0 = load_a_frag_bf16(hin, HID, m0a, kk * 32, lane);
      v16bf a1 = load_a_frag_bf16(hin, HID, m0b, kk * 32, lane);
      acc0 = wmma_bf16(a0, b, acc0);
      acc1 = wmma_bf16(a1, b, acc1);
    }

    // bias + tanh + store next state (D layout: lane = column, VGPR i = row)
    {
      const int n  = n0 + (lane & 15);
      const int ma = m0a + ((lane & 16) ? 8 : 0);
      const int mb = m0b + ((lane & 16) ? 8 : 0);
#pragma unroll
      for (int i = 0; i < 8; ++i) {
        hout[(size_t)(ma + i) * HID + n] = (__bf16)tanhf(acc0[i] + bias_h);
        hout[(size_t)(mb + i) * HID + n] = (__bf16)tanhf(acc1[i] + bias_h);
      }
    }

    // grid-wide barrier: monotonic counter, release/acquire at device scope
    __syncthreads();
    if (threadIdx.x == 0) {
      __threadfence();
      __hip_atomic_fetch_add(counter, 1u, __ATOMIC_RELEASE,
                             __HIP_MEMORY_SCOPE_AGENT);
      const unsigned target = (unsigned)NWG * (unsigned)(t + 1);
      while (__hip_atomic_load(counter, __ATOMIC_ACQUIRE,
                               __HIP_MEMORY_SCOPE_AGENT) < target) {
        __builtin_amdgcn_s_sleep(0);
      }
      __threadfence();
    }
    __syncthreads();
  }

  // ---- final GEMM: y = h_final @ W_ho^T + b_ho  (256 tiles, 1 per wave) ----
  const bf16_t* hf = h0;                       // TSTEPS even -> last write hit h0
  const int idx = wg * WAVES + wave;           // 0..255
  const int m0  = (idx & 7) * 16;              // 8 batch tiles
  const int o0  = (idx >> 3) * 16;             // 32 output tiles

  v8f acc = {};
#pragma unroll
  for (int kk = 0; kk < HID / 32; ++kk) {
    v16bf b = load_b_frag(Who, HID, o0, kk * 32, lane);
    v16bf a = load_a_frag_bf16(hf, HID, m0, kk * 32, lane);
    acc = wmma_bf16(a, b, acc);
  }
  const int o  = o0 + (lane & 15);
  const int mb = m0 + ((lane & 16) ? 8 : 0);
  const float bias = b_ho[o];
#pragma unroll
  for (int i = 0; i < 8; ++i) {
    out[(size_t)(mb + i) * OUTSZ + o] = acc[i] + bias;
  }
}

// ---------------------------------------------------------------------------

extern "C" void kernel_launch(void* const* d_in, const int* in_sizes, int n_in,
                              void* d_out, int out_size, void* d_ws, size_t ws_size,
                              hipStream_t stream) {
  const float* x     = (const float*)d_in[0];  // (128, 1024, 512)
  const float* Wih_f = (const float*)d_in[1];  // (1024, 1536)
  const float* b_ih  = (const float*)d_in[2];  // (1024,)
  const float* Who_f = (const float*)d_in[3];  // (512, 1024)
  const float* b_ho  = (const float*)d_in[4];  // (512,)
  float* out = (float*)d_out;                  // (128, 512)

  // workspace layout (all 256B aligned): counter | Wih_bf | Who_bf | h double buf
  uintptr_t base = (uintptr_t)d_ws;
  unsigned* counter = (unsigned*)base;
  bf16_t* Wih = (bf16_t*)(base + 256);
  bf16_t* Who = (bf16_t*)(base + 256 + (size_t)HID * KTOT * 2);
  bf16_t* hbuf = (bf16_t*)(base + 256 + (size_t)HID * KTOT * 2
                                     + (size_t)OUTSZ * HID * 2);

  (void)hipMemsetAsync((void*)counter, 0, 256, stream);
  rnn_prologue<<<1024, 256, 0, stream>>>(Wih_f, Who_f, Wih, Who, hbuf);
  rnn_persistent<<<NWG, WAVES * 32, 0, stream>>>(x, b_ih, b_ho, Wih, Who,
                                                 hbuf, out, counter);
}
